// Network_nerf_12223476924999
// MI455X (gfx1250) — matrix operand order repaired and checked by reading, not compiled, not run on hardware
//
#include <hip/hip_runtime.h>
#include <hip/hip_bf16.h>
#include <math.h>

typedef __attribute__((ext_vector_type(16))) _Float16 v16h;
typedef __attribute__((ext_vector_type(8)))  _Float16 v8h;
typedef __attribute__((ext_vector_type(8)))  float    v8f;
typedef int v4i __attribute__((vector_size(16)));   // matches builtin param pointee

// ---------------------------------------------------------------------------
// gfx1250 async global->LDS copy (16B per lane), with portable fallback.
// Builtin signature (from hipcc diagnostic): (v4i AS1*, v4i AS3*, imm, imm).
// ---------------------------------------------------------------------------
#if defined(__has_builtin)
#if __has_builtin(__builtin_amdgcn_global_load_async_to_lds_b128)
#define HAS_ASYNC_LDS 1
#endif
#endif
#ifndef HAS_ASYNC_LDS
#define HAS_ASYNC_LDS 0
#endif

__device__ __forceinline__ void copy16B_g2lds(const _Float16* g, _Float16* l) {
#if HAS_ASYNC_LDS
  __builtin_amdgcn_global_load_async_to_lds_b128(
      (__attribute__((address_space(1))) v4i*)(g),
      (__attribute__((address_space(3))) v4i*)(l), 0, 0);
#else
  *(v8h*)l = *(const v8h*)g;
#endif
}
__device__ __forceinline__ void async_copy_fence() {
#if HAS_ASYNC_LDS
#if __has_builtin(__builtin_amdgcn_s_wait_asynccnt)
  __builtin_amdgcn_s_wait_asynccnt(0);
#else
  asm volatile("s_wait_asynccnt 0x0" ::: "memory");
#endif
#endif
}

// ---------------------------------------------------------------------------
// WMMA fragment loaders (wave32, per CDNA5 ISA 7.12.2 VGPR layouts)
// A 16x32 f16 from LDS [m][k] (row stride multiple of 8 halves, 16B-aligned):
// lane = hi*16+m needs k = hi*8 + [0..7] and 16 + hi*8 + [0..7]  -> 2x b128.
// ---------------------------------------------------------------------------
__device__ __forceinline__ v16h load_A_frag(const _Float16* lds, int ldk) {
  const int lane = threadIdx.x & 31;
  const int m = lane & 15, hi = lane >> 4;
  const _Float16* row = lds + m * ldk + hi * 8;
  const v8h lo = *(const v8h*)(row);
  const v8h hg = *(const v8h*)(row + 16);
  v16h a;
#pragma unroll
  for (int e = 0; e < 8; ++e) { a[e] = lo[e]; a[8 + e] = hg[e]; }
  return a;
}

// B staged fragment-major: blk[lane][e] = B[k=(e<8 ? (lane>>4)*8+e : 16+(lane>>4)*8+e-8)][n=lane&15]; 32B/lane.
__device__ __forceinline__ v16h load_B_frag(const _Float16* lds) {
  const int lane = threadIdx.x & 31;
  return *(const v16h*)(lds + lane * 16);
}
// C/D 16x16 f32: vgpr r -> (m = r + 8*(lane>>4), n = lane&15)

__device__ __forceinline__ float fast_sigmoid(float x) {
  return __fdividef(1.f, 1.f + __expf(-x));
}
__device__ __forceinline__ float fast_tanh(float x) {
  x = fminf(fmaxf(x, -12.f), 12.f);
  const float t = __expf(2.f * x);
  return __fdividef(t - 1.f, t + 1.f);
}
__device__ __forceinline__ float selu(float x) {
  return 1.0507009873554805f * (x > 0.f ? x : 1.6732632423543772f * (__expf(x) - 1.f));
}

// ---------------------------------------------------------------------------
// Weight prep: OIHW f32 -> fragment-major f16 chunks [ch][nt][lane][16].
// ---------------------------------------------------------------------------
template <int CIN, int COUT, int KSZ>
__global__ void prep_conv_w(const float* __restrict__ W, _Float16* __restrict__ dst) {
  constexpr int KTOT = CIN * KSZ * KSZ;
  constexpr int KCH  = (KTOT + 31) / 32;
  constexpr int NT   = (COUT > 16) ? 2 : 1;
  const int i = blockIdx.x * blockDim.x + threadIdx.x;
  if (i >= KCH * NT * 512) return;
  const int ch  = i / (NT * 512);
  const int rem = i - ch * (NT * 512);
  const int nt2 = rem >> 9;
  const int l   = (rem >> 4) & 31;
  const int q   = rem & 15;
  const int n   = nt2 * 16 + (l & 15);
  const int k   = (q < 8) ? ((l >> 4) << 3) + q : 16 + ((l >> 4) << 3) + (q - 8);
  const int kk  = ch * 32 + k;
  float v = 0.f;
  if (kk < KTOT && n < COUT) {
    const int ci = kk % CIN;
    const int t  = kk / CIN;
    const int kx = t % KSZ, ky = t / KSZ;
    v = W[((n * CIN + ci) * KSZ + ky) * KSZ + kx];   // OIHW
  }
  dst[i] = (_Float16)v;
}

// MLP weights: w1 cat (4 tiles) then w2 pn/ps (2 tiles, zero padded): 3072 halves
__global__ void prep_mlp_w(const float* __restrict__ pn_w1, const float* __restrict__ ps_w1,
                           const float* __restrict__ pn_w2, const float* __restrict__ ps_w2,
                           _Float16* __restrict__ dst) {
  const int i = blockIdx.x * blockDim.x + threadIdx.x;
  if (i >= 3072) return;
  float v = 0.f;
  if (i < 2048) {
    const int ntt = i >> 9, l = (i >> 4) & 31, q = i & 15;
    const int k = (q < 8) ? ((l >> 4) << 3) + q : 16 + ((l >> 4) << 3) + (q - 8);
    const int j = ntt * 16 + (l & 15);
    v = (j < 32) ? pn_w1[k * 32 + j] : ps_w1[k * 32 + (j - 32)];
  } else {
    const int e = i - 2048;
    const int tt = e >> 9, l = (e >> 4) & 31, q = e & 15;
    const int k = (q < 8) ? ((l >> 4) << 3) + q : 16 + ((l >> 4) << 3) + (q - 8);
    const int n = l & 15;
    if (tt == 0) { if (n < 3) v = pn_w2[k * 3 + n]; }
    else         { if (n < 7) v = ps_w2[k * 7 + n]; }
  }
  dst[i] = (_Float16)v;
}

// ---------------------------------------------------------------------------
// Implicit-GEMM conv + fused BN-apply(prev)+LeakyReLU on input read.
// MODE 0: first layer (float NHWC x, x/2-1), emit BN stats
// MODE 1: mid layer   (raw f16 NHWC + scale/shift + lrelu), emit BN stats
// MODE 2: top layer   (like MODE1, +bias, no stats)
// Tile: 64 output pixels x COUT; waves = 4 Mtiles x NT Ntiles.
// Weights arrive pre-swizzled f16; whole layer block async-copied to LDS once.
// ---------------------------------------------------------------------------
template <int CIN, int COUT, int KSZ, int STRIDE, int PAD, int MODE,
          int HIN, int WIN, int HOUT, int WOUT>
__global__ __launch_bounds__(256) void conv_wmma(
    const void* __restrict__ inp, const float* __restrict__ in_ss,
    const _Float16* __restrict__ Wf, const float* __restrict__ bias,
    _Float16* __restrict__ out, float* __restrict__ stats)
{
  constexpr int KTOT = CIN * KSZ * KSZ;
  constexpr int KCH  = (KTOT + 31) / 32;
  constexpr int NT   = (COUT > 16) ? 2 : 1;
  constexpr int HW   = HOUT * WOUT;
  constexpr int WHALVES = KCH * NT * 512;

  __shared__ __align__(16) _Float16 At[64][32];
  __shared__ __align__(32) _Float16 Wlds[KCH][NT][32][16];
  __shared__ float ssc[32], ssh[32];
  __shared__ float bsum[32], bss[32];

  const int tid  = threadIdx.x;
  const int pix0 = blockIdx.x * 64;

  // async-copy full pre-swizzled weight block into LDS
  {
    _Float16* wl = &Wlds[0][0][0][0];
    for (int e = tid * 8; e < WHALVES; e += (int)blockDim.x * 8)
      copy16B_g2lds(Wf + e, wl + e);
  }
  if (MODE != 0)
    for (int c = tid; c < CIN; c += blockDim.x) { ssc[c] = in_ss[c]; ssh[c] = in_ss[32 + c]; }
  if (MODE != 2)
    for (int c = tid; c < COUT; c += blockDim.x) { bsum[c] = 0.f; bss[c] = 0.f; }
  async_copy_fence();
  __syncthreads();

  const int wv = tid >> 5;
  const int mt = wv & 3, nt = wv >> 2;
  v8f acc = {};

  for (int ch = 0; ch < KCH; ++ch) {
    // ---- stage im2col A tile: k-order = (ky*KSZ+kx)*CIN + ci ----
    if (MODE != 0) {
      // CIN multiple of 8 -> aligned groups of 8 k are one (ky,kx), contiguous ci
      for (int e = tid; e < 64 * 4; e += blockDim.x) {
        const int p = e >> 2, kb = (e & 3) << 3;
        const int kk = ch * 32 + kb;
        v8h vv = {};
        if (kk < KTOT) {
          const int ci = kk % CIN;
          const int t  = kk / CIN;
          const int kx = t % KSZ, ky = t / KSZ;
          const int g  = pix0 + p;
          const int b  = g / HW;
          const int r  = g - b * HW;
          const int oy = r / WOUT, ox = r - (r / WOUT) * WOUT;
          const int iy = oy * STRIDE - PAD + ky, ix = ox * STRIDE - PAD + kx;
          if (iy >= 0 && iy < HIN && ix >= 0 && ix < WIN) {
            const _Float16* src =
                (const _Float16*)inp + ((size_t)(b * HIN + iy) * WIN + ix) * CIN + ci;
            const v8h raw = *(const v8h*)src;
#pragma unroll
            for (int q = 0; q < 8; ++q) {
              const float a = (float)raw[q] * ssc[ci + q] + ssh[ci + q];
              vv[q] = (_Float16)((a >= 0.f) ? a : 0.01f * a);
            }
          }
        }
        *(v8h*)(&At[p][kb]) = vv;
      }
    } else {
      for (int e = tid; e < 64 * 32; e += blockDim.x) {
        const int p = e >> 5, k = e & 31;
        const int kk = ch * 32 + k;
        float v = 0.f;
        if (kk < KTOT) {
          const int ci = kk % CIN;
          const int t  = kk / CIN;
          const int kx = t % KSZ, ky = t / KSZ;
          const int g  = pix0 + p;
          const int b  = g / HW;
          const int r  = g - b * HW;
          const int oy = r / WOUT, ox = r - (r / WOUT) * WOUT;
          const int iy = oy * STRIDE - PAD + ky, ix = ox * STRIDE - PAD + kx;
          if (iy >= 0 && iy < HIN && ix >= 0 && ix < WIN)
            v = ((const float*)inp)[((size_t)(b * HIN + iy) * WIN + ix) * CIN + ci] * 0.5f - 1.0f;
        }
        At[p][k] = (_Float16)v;
      }
    }
    __syncthreads();
    const v16h a = load_A_frag(&At[mt * 16][0], 32);
    const v16h b = load_B_frag(&Wlds[ch][nt][0][0]);
    acc = __builtin_amdgcn_wmma_f32_16x16x32_f16(false, a, false, b, (short)0, acc, false, false);
    __syncthreads();
  }

  // ---- epilogue: raw output + per-channel stats ----
  const int lane = tid & 31;
  const int n = nt * 16 + (lane & 15);
#pragma unroll
  for (int r = 0; r < 8; ++r) {
    const int m = mt * 16 + r + ((lane >> 4) << 3);
    const long long g = pix0 + m;
    float v = acc[r];
    if (n < COUT) {
      if (MODE == 2) v += bias[n];
      out[g * COUT + n] = (_Float16)v;
      if (MODE != 2) { atomicAdd(&bsum[n], v); atomicAdd(&bss[n], v * v); }
    }
  }
  if (MODE != 2) {
    __syncthreads();
    for (int c = tid; c < COUT; c += blockDim.x) {
      atomicAdd(&stats[c],      bsum[c]);
      atomicAdd(&stats[32 + c], bss[c]);
    }
  }
}

// fold BN statistics + gamma/beta into per-channel (scale, shift)
__global__ void bn_finalize(const float* __restrict__ stats,
                            const float* __restrict__ gamma, const float* __restrict__ beta,
                            float* __restrict__ ss, float invN, int C) {
  const int c = threadIdx.x;
  if (c < C) {
    const float mean = stats[c] * invN;
    const float var  = stats[32 + c] * invN - mean * mean;
    const float sc   = rsqrtf(var + 1e-5f) * gamma[c];
    ss[c]      = sc;
    ss[32 + c] = beta[c] - mean * sc;
  }
}

__global__ void zero_stats(float* p, int n) {
  const int i = blockIdx.x * blockDim.x + threadIdx.x;
  if (i < n) p[i] = 0.f;
}

// ---------------------------------------------------------------------------
// Fused: x4 bilinear upsample (align_corners) -> MLPs pn/ps (WMMA) -> GGX.
// 256 threads = 8 waves; each wave owns 32 pixels (2 WMMA M-tiles).
// ---------------------------------------------------------------------------
__global__ __launch_bounds__(256) void mlp_render(
    const _Float16* __restrict__ feat,   // [B][128][128][32] f16
    const _Float16* __restrict__ wmlp_g, // pre-swizzled: w1f(2048) + w2f(1024)
    const float* __restrict__ wi, const float* __restrict__ wo,
    const float* __restrict__ light,
    const float* __restrict__ pn_b1, const float* __restrict__ pn_b2,
    const float* __restrict__ ps_b1, const float* __restrict__ ps_b2,
    float* __restrict__ imgs, float* __restrict__ data)
{
  const int tid = threadIdx.x, lane = tid & 31, wv = tid >> 5;

  __shared__ __align__(32) _Float16 wmlp[3072];       // w1f[4][512] | w2f[2][512]
  __shared__ float b1c[64];
  __shared__ float b2c[32];
  __shared__ __align__(16) _Float16 ftile[8][32][40]; // feat tile; reused as output tile
  __shared__ __align__(16) _Float16 htile[8][32][72]; // hidden tile (64 wide + pad)

  for (int e = tid * 8; e < 3072; e += 2048)
    copy16B_g2lds(wmlp_g + e, wmlp + e);
  for (int e = tid; e < 64; e += 256) b1c[e] = (e < 32) ? pn_b1[e] : ps_b1[e - 32];
  for (int e = tid; e < 32; e += 256) {
    float v = 0.f;
    if (e < 3) v = pn_b2[e];
    else if (e >= 16 && e < 23) v = ps_b2[e - 16];
    b2c[e] = v;
  }

  // --- bilinear feature gather: one pixel per lane ---
  const int p  = blockIdx.x * 256 + wv * 32 + lane;   // flattened over B*512*512
  const int b  = p >> 18;
  const int rp = p & 262143;
  const int oy = rp >> 9, ox = rp & 511;

  __builtin_prefetch(wi + (size_t)p * 3, 0, 0);       // global_prefetch_b8
  __builtin_prefetch(wo + (size_t)p * 3, 0, 0);

  const float s = 127.0f / 511.0f;
  const float yi = oy * s, xi = ox * s;
  const int y0 = (int)floorf(yi), x0 = (int)floorf(xi);
  const int y1 = (y0 + 1 < 127) ? y0 + 1 : 127;
  const int x1 = (x0 + 1 < 127) ? x0 + 1 : 127;
  const float wy = yi - (float)y0, wx = xi - (float)x0;
  const _Float16* f00 = feat + (((size_t)(b * 128 + y0) * 128 + x0) << 5);
  const _Float16* f01 = feat + (((size_t)(b * 128 + y0) * 128 + x1) << 5);
  const _Float16* f10 = feat + (((size_t)(b * 128 + y1) * 128 + x0) << 5);
  const _Float16* f11 = feat + (((size_t)(b * 128 + y1) * 128 + x1) << 5);
#pragma unroll
  for (int cb = 0; cb < 4; ++cb) {
    const v8h a00 = *(const v8h*)(f00 + cb * 8);
    const v8h a01 = *(const v8h*)(f01 + cb * 8);
    const v8h a10 = *(const v8h*)(f10 + cb * 8);
    const v8h a11 = *(const v8h*)(f11 + cb * 8);
    v8h r8;
#pragma unroll
    for (int q = 0; q < 8; ++q) {
      const float top = (float)a00[q] * (1.f - wx) + (float)a01[q] * wx;
      const float bot = (float)a10[q] * (1.f - wx) + (float)a11[q] * wx;
      r8[q] = (_Float16)(top * (1.f - wy) + bot * wy);
    }
    *(v8h*)(&ftile[wv][lane][cb * 8]) = r8;
  }
  async_copy_fence();
  __syncthreads();

  // --- layer 1: feat(32) @ [w1_pn | w1_ps](32x64), SELU ---
  for (int mt = 0; mt < 2; ++mt) {
    const v16h a = load_A_frag(&ftile[wv][mt * 16][0], 40);
#pragma unroll
    for (int ntt = 0; ntt < 4; ++ntt) {
      const v16h bf = load_B_frag(&wmlp[ntt * 512]);
      v8f c = {};
      c = __builtin_amdgcn_wmma_f32_16x16x32_f16(false, a, false, bf, (short)0, c, false, false);
      const int n = ntt * 16 + (lane & 15);
#pragma unroll
      for (int r = 0; r < 8; ++r) {
        const int m = mt * 16 + r + ((lane >> 4) << 3);
        htile[wv][m][n] = (_Float16)selu(c[r] + b1c[n]);
      }
    }
  }
  __syncthreads();

  // --- layer 2: h_pn(32)@w2_pn(pad16) tanh ; h_ps(32)@w2_ps(pad16) sigmoid ---
  for (int mt = 0; mt < 2; ++mt) {
    const v16h apn = load_A_frag(&htile[wv][mt * 16][0],  72);
    const v16h aps = load_A_frag(&htile[wv][mt * 16][32], 72);
    const v16h bpn = load_B_frag(&wmlp[2048]);
    const v16h bps = load_B_frag(&wmlp[2560]);
    v8f cpn = {}, cps = {};
    cpn = __builtin_amdgcn_wmma_f32_16x16x32_f16(false, apn, false, bpn, (short)0, cpn, false, false);
    cps = __builtin_amdgcn_wmma_f32_16x16x32_f16(false, aps, false, bps, (short)0, cps, false, false);
    const int n = lane & 15;
#pragma unroll
    for (int r = 0; r < 8; ++r) {
      const int m = mt * 16 + r + ((lane >> 4) << 3);
      if (n < 3) ftile[wv][m][n]      = (_Float16)fast_tanh(cpn[r] + b2c[n]);
      if (n < 7) ftile[wv][m][16 + n] = (_Float16)fast_sigmoid(cps[r] + b2c[16 + n]);
    }
  }
  __syncthreads();

  // --- per-pixel epilogue: normalize / clip / assemble data / GGX ---
  float nx = (float)ftile[wv][lane][0], ny = (float)ftile[wv][lane][1], nz = (float)ftile[wv][lane][2];
  const float nl = fmaxf(sqrtf(nx * nx + ny * ny + nz * nz), 1e-12f);
  nx /= nl; ny /= nl; nz /= nl;
  float dif[3], sp[3];
#pragma unroll
  for (int c = 0; c < 3; ++c) {
    dif[c] = fminf(fmaxf((float)ftile[wv][lane][16 + c], 0.001f), 1.f);
    sp[c]  = fminf(fmaxf((float)ftile[wv][lane][20 + c], 0.f),    1.f);
  }
  const float rough = fminf(fmaxf((float)ftile[wv][lane][19], 0.001f), 1.f);

  const size_t dbase = (size_t)p * 12;
  data[dbase + 0] = nx; data[dbase + 1] = ny; data[dbase + 2] = nz;
  data[dbase + 3] = dif[0]; data[dbase + 4] = dif[1]; data[dbase + 5] = dif[2];
  data[dbase + 6] = rough;  data[dbase + 7] = rough;  data[dbase + 8] = rough;
  data[dbase + 9] = sp[0];  data[dbase + 10] = sp[1]; data[dbase + 11] = sp[2];

  float wix = wi[(size_t)p * 3 + 0], wiy = wi[(size_t)p * 3 + 1], wiz = wi[(size_t)p * 3 + 2];
  float wox = wo[(size_t)p * 3 + 0], woy = wo[(size_t)p * 3 + 1], woz = wo[(size_t)p * 3 + 2];
  const float il = fmaxf(sqrtf(wix * wix + wiy * wiy + wiz * wiz), 1e-12f); wix /= il; wiy /= il; wiz /= il;
  const float ol = fmaxf(sqrtf(wox * wox + woy * woy + woz * woz), 1e-12f); wox /= ol; woy /= ol; woz /= ol;
  float hx = (wix + wox) * 0.5f, hy = (wiy + woy) * 0.5f, hz = (wiz + woz) * 0.5f;
  const float hl = fmaxf(sqrtf(hx * hx + hy * hy + hz * hz), 1e-12f); hx /= hl; hy /= hl; hz /= hl;

  const float NdotH = fmaxf(nx * hx + ny * hy + nz * hz, 0.f);
  const float NdotL = fmaxf(nx * wix + ny * wiy + nz * wiz, 0.f);
  const float NdotV = fmaxf(nx * wox + ny * woy + nz * woz, 0.f);
  const float VdotH = fmaxf(wox * hx + woy * hy + woz * hz, 0.f);

  const float INV_PI = 0.3183098861837907f;
  const float alpha = rough * rough, a2 = alpha * alpha;
  const float underD = __fdividef(1.f, fmaxf(NdotH * NdotH * (a2 - 1.f) + 1.f, 0.001f));
  const float D = a2 * underD * underD * INV_PI;
  const float kk = fmaxf(alpha * 0.5f, 0.001f);
  const float G = __fdividef(1.f, fmaxf(NdotL * (1.f - kk) + kk, 0.001f)) *
                  __fdividef(1.f, fmaxf(NdotV * (1.f - kk) + kk, 0.001f));
  const float Fexp = exp2f((-5.55473f * VdotH - 6.98316f) * VdotH);
#pragma unroll
  for (int c = 0; c < 3; ++c) {
    const float F  = sp[c] + (1.f - sp[c]) * Fexp;
    const float fd = dif[c] * (1.f - sp[c]) * INV_PI;
    const float fs = D * F * G * 0.25f;
    imgs[(size_t)p * 3 + c] = (fd + fs) * NdotL * light[b * 3 + c];
  }
}

// ---------------------------------------------------------------------------
extern "C" void kernel_launch(void* const* d_in, const int* in_sizes, int n_in,
                              void* d_out, int out_size, void* d_ws, size_t ws_size,
                              hipStream_t stream) {
  (void)in_sizes; (void)n_in; (void)out_size; (void)ws_size;
  const float* wi    = (const float*)d_in[0];
  const float* wo    = (const float*)d_in[1];
  const float* light = (const float*)d_in[2];
  const float* x     = (const float*)d_in[3];
  auto Wp = [&](int i) { return (const float*)d_in[4 + 3 * i]; };
  auto Gp = [&](int i) { return (const float*)d_in[5 + 3 * i]; };
  auto Bp = [&](int i) { return (const float*)d_in[6 + 3 * i]; };
  const float* top_w = (const float*)d_in[28];
  const float* top_b = (const float*)d_in[29];
  const float* pn_w1 = (const float*)d_in[30];
  const float* pn_b1 = (const float*)d_in[31];
  const float* pn_w2 = (const float*)d_in[32];
  const float* pn_b2 = (const float*)d_in[33];
  const float* ps_w1 = (const float*)d_in[34];
  const float* ps_b1 = (const float*)d_in[35];
  const float* ps_w2 = (const float*)d_in[36];
  const float* ps_b2 = (const float*)d_in[37];

  // activation ping-pong lives inside d_out (dead until final kernel)
  _Float16* buf0 = (_Float16*)d_out;
  _Float16* buf1 = (_Float16*)((char*)d_out + (size_t)(24u << 20));
  _Float16* feat  = (_Float16*)d_ws;                              // 2 MB
  float*    stats = (float*)((char*)d_ws + (size_t)(2u << 20));   // 8 layers * 64
  float*    ss    = stats + 512;                                  // 8 layers * 64
  _Float16* wbuf  = (_Float16*)((char*)d_ws + (size_t)(2u << 20) + 8192);
  float* imgs = (float*)d_out;
  float* data = (float*)d_out + (size_t)4 * 512 * 512 * 3;

  // fragment-major weight block offsets (halves)
  const int wo0 = 0, wo1 = 512, wo2 = 2048, wo3 = 5632, wo4 = 8192;
  const int wo5 = 10752, wo6 = 24064, wo7 = 33280, wot = 42496, wom = 43520;

  zero_stats<<<4, 256, 0, stream>>>(stats, 1024);

  prep_conv_w<3, 8, 3><<<2, 256, 0, stream>>>(Wp(0), wbuf + wo0);
  prep_conv_w<8, 8, 3><<<6, 256, 0, stream>>>(Wp(1), wbuf + wo1);
  prep_conv_w<8, 16, 5><<<14, 256, 0, stream>>>(Wp(2), wbuf + wo2);
  prep_conv_w<16, 16, 3><<<10, 256, 0, stream>>>(Wp(3), wbuf + wo3);
  prep_conv_w<16, 16, 3><<<10, 256, 0, stream>>>(Wp(4), wbuf + wo4);
  prep_conv_w<16, 32, 5><<<52, 256, 0, stream>>>(Wp(5), wbuf + wo5);
  prep_conv_w<32, 32, 3><<<36, 256, 0, stream>>>(Wp(6), wbuf + wo6);
  prep_conv_w<32, 32, 3><<<36, 256, 0, stream>>>(Wp(7), wbuf + wo7);
  prep_conv_w<32, 32, 1><<<4, 256, 0, stream>>>(top_w, wbuf + wot);
  prep_mlp_w<<<12, 256, 0, stream>>>(pn_w1, ps_w1, pn_w2, ps_w2, wbuf + wom);

  const float iN0 = 1.f / 1048576.f, iN1 = 1.f / 262144.f, iN2 = 1.f / 65536.f;

  conv_wmma<3, 8, 3, 1, 1, 0, 512, 512, 512, 512><<<16384, 128, 0, stream>>>(x, nullptr, wbuf + wo0, nullptr, buf0, stats + 0);
  bn_finalize<<<1, 32, 0, stream>>>(stats + 0, Gp(0), Bp(0), ss + 0, iN0, 8);
  conv_wmma<8, 8, 3, 1, 1, 1, 512, 512, 512, 512><<<16384, 128, 0, stream>>>(buf0, ss + 0, wbuf + wo1, nullptr, buf1, stats + 64);
  bn_finalize<<<1, 32, 0, stream>>>(stats + 64, Gp(1), Bp(1), ss + 64, iN0, 8);
  conv_wmma<8, 16, 5, 2, 2, 1, 512, 512, 256, 256><<<4096, 128, 0, stream>>>(buf1, ss + 64, wbuf + wo2, nullptr, buf0, stats + 128);
  bn_finalize<<<1, 32, 0, stream>>>(stats + 128, Gp(2), Bp(2), ss + 128, iN1, 16);
  conv_wmma<16, 16, 3, 1, 1, 1, 256, 256, 256, 256><<<4096, 128, 0, stream>>>(buf0, ss + 128, wbuf + wo3, nullptr, buf1, stats + 192);
  bn_finalize<<<1, 32, 0, stream>>>(stats + 192, Gp(3), Bp(3), ss + 192, iN1, 16);
  conv_wmma<16, 16, 3, 1, 1, 1, 256, 256, 256, 256><<<4096, 128, 0, stream>>>(buf1, ss + 192, wbuf + wo4, nullptr, buf0, stats + 256);
  bn_finalize<<<1, 32, 0, stream>>>(stats + 256, Gp(4), Bp(4), ss + 256, iN1, 16);
  conv_wmma<16, 32, 5, 2, 2, 1, 256, 256, 128, 128><<<1024, 256, 0, stream>>>(buf0, ss + 256, wbuf + wo5, nullptr, buf1, stats + 320);
  bn_finalize<<<1, 32, 0, stream>>>(stats + 320, Gp(5), Bp(5), ss + 320, iN2, 32);
  conv_wmma<32, 32, 3, 1, 1, 1, 128, 128, 128, 128><<<1024, 256, 0, stream>>>(buf1, ss + 320, wbuf + wo6, nullptr, buf0, stats + 384);
  bn_finalize<<<1, 32, 0, stream>>>(stats + 384, Gp(6), Bp(6), ss + 384, iN2, 32);
  conv_wmma<32, 32, 3, 1, 1, 1, 128, 128, 128, 128><<<1024, 256, 0, stream>>>(buf0, ss + 384, wbuf + wo7, nullptr, buf1, stats + 448);
  bn_finalize<<<1, 32, 0, stream>>>(stats + 448, Gp(7), Bp(7), ss + 448, iN2, 32);
  // top 1x1 conv: BN(c2c)+lrelu on read, +bias, f16 NHWC feature map
  conv_wmma<32, 32, 1, 1, 0, 2, 128, 128, 128, 128><<<1024, 256, 0, stream>>>(buf1, ss + 448, wbuf + wot, top_b, feat, nullptr);

  mlp_render<<<4096, 256, 0, stream>>>(feat, wbuf + wom, wi, wo, light,
                                       pn_b1, pn_b2, ps_b1, ps_b2,
                                       imgs, data);
}
